// Attention_54107997995066
// MI455X (gfx1250) — matrix operand order repaired and checked by reading, not compiled, not run on hardware
//
#include <hip/hip_runtime.h>
#include <hip/hip_bf16.h>

// ---- CDNA5 vector types for WMMA ----
typedef __attribute__((ext_vector_type(16))) __bf16 bf16x16;
typedef __attribute__((ext_vector_type(8)))  __bf16 bf16x8;
typedef __attribute__((ext_vector_type(4)))  __bf16 bf16x4;
typedef __attribute__((ext_vector_type(8)))  float  f32x8;

#define T_LEN 2048
#define B_SZ  32
#define LIS_D 1024
#define ATT_D 512
#define MB    128          // t-rows per workgroup (8 waves x 16)
#define THREADS 256

// -----------------------------------------------------------------------------
// Kernel 1: transpose-convert W_score / W_value (fp32 [k][a]) -> bf16 [a][k],
// and zero the context region of d_out (fallback path accumulates atomically).
// -----------------------------------------------------------------------------
__global__ void prep_kernel(const float* __restrict__ Ws, const float* __restrict__ Wv,
                            __bf16* __restrict__ WtS, __bf16* __restrict__ WtV,
                            float* __restrict__ ctxOut) {
    int idx = blockIdx.x * THREADS + threadIdx.x;   // 0 .. 512*1024-1
    int a = idx & (ATT_D - 1);
    int k = idx >> 9;
    WtS[((size_t)a << 10) + k] = (__bf16)Ws[((size_t)k << 9) + a];
    WtV[((size_t)a << 10) + k] = (__bf16)Wv[((size_t)k << 9) + a];
    if (idx < B_SZ * ATT_D) ctxOut[idx] = 0.0f;
}

// -----------------------------------------------------------------------------
// Kernel 2: query = relu(speller @ W_proj + b_proj)   [32 x 512], tiny.
// -----------------------------------------------------------------------------
__global__ void query_kernel(const float* __restrict__ sp, const float* __restrict__ Wp,
                             const float* __restrict__ bp, float* __restrict__ q) {
    int b = blockIdx.x;
    int a = threadIdx.x;                 // 512 threads
    float acc = bp[a];
    const float* sb = sp + ((size_t)b << 10);
    #pragma unroll 8
    for (int d = 0; d < LIS_D; ++d)
        acc = fmaf(sb[d], Wp[((size_t)d << 9) + a], acc);
    q[((size_t)b << 9) + a] = fmaxf(acc, 0.0f);
}

// -----------------------------------------------------------------------------
// WMMA helper
// -----------------------------------------------------------------------------
__device__ __forceinline__ f32x8 wmma_bf16(bf16x16 a, bf16x16 b, f32x8 c) {
    return __builtin_amdgcn_wmma_f32_16x16x32_bf16(
        /*neg_a=*/false, a, /*neg_b=*/false, b,
        /*c_mod=*/(short)0, c, /*reuse_a=*/false, /*reuse_b=*/false);
}

// Stage x[b, t0..t0+127, 0..1023] (fp32, [T][B][LIS] layout) into LDS as bf16.
__device__ __forceinline__ void stage_A(const float* __restrict__ xb, int t0,
                                        __bf16* __restrict__ As) {
    for (int i = threadIdx.x; i < MB * 256; i += THREADS) {
        int m  = i >> 8;
        int d4 = (i & 255) << 2;
        const float4 v = *(const float4*)(xb + (size_t)(t0 + m) * (B_SZ * LIS_D) + d4);
        bf16x4 h = { (__bf16)v.x, (__bf16)v.y, (__bf16)v.z, (__bf16)v.w };
        *(bf16x4*)(As + m * LIS_D + d4) = h;
    }
}

// -----------------------------------------------------------------------------
// FUSED kernel: one pass over x computes BOTH
//   keys -> scores[b,t]  and  vals (relu'd, bf16) -> VALS[b][a][t]
// Two independent WMMA chains share the A-fragments (half the ds traffic,
// double the WMMA ILP). x is read from HBM exactly once.
// -----------------------------------------------------------------------------
__global__ __launch_bounds__(THREADS)
void scores_vals_kernel(const float* __restrict__ listener,
                        const __bf16* __restrict__ WtS, const __bf16* __restrict__ WtV,
                        const float* __restrict__ bscore, const float* __restrict__ bvalue,
                        const float* __restrict__ Q,
                        float* __restrict__ SC, __bf16* __restrict__ VALS) {
    extern __shared__ char smem[];
    __bf16* As = (__bf16*)smem;

    const int b  = blockIdx.x >> 4;
    const int t0 = (blockIdx.x & 15) * MB;
    const float* xb = listener + ((size_t)b << 10);

    stage_A(xb, t0, As);
    __syncthreads();

    const int wave = threadIdx.x >> 5;
    const int lane = threadIdx.x & 31;
    const int g    = lane >> 4;
    const int ln   = lane & 15;
    const __bf16* arow = As + (wave * 16 + ln) * LIS_D;
    const int trow = t0 + wave * 16 + g * 8;   // first of this lane's 8 C-rows

    float sAcc[8] = {0.f, 0.f, 0.f, 0.f, 0.f, 0.f, 0.f, 0.f};

    for (int a0 = 0; a0 < ATT_D; a0 += 16) {
        f32x8 accK = {0.f, 0.f, 0.f, 0.f, 0.f, 0.f, 0.f, 0.f};
        f32x8 accV = {0.f, 0.f, 0.f, 0.f, 0.f, 0.f, 0.f, 0.f};
        const __bf16* bcolS = WtS + ((size_t)(a0 + ln) << 10) + g * 16;
        const __bf16* bcolV = WtV + ((size_t)(a0 + ln) << 10) + g * 16;
        #pragma unroll 4
        for (int kk = 0; kk < LIS_D; kk += 32) {
            bf16x8 lo = *(const bf16x8*)(arow + kk + g * 16);
            bf16x8 hi = *(const bf16x8*)(arow + kk + g * 16 + 8);
            bf16x16 af = __builtin_shufflevector(lo, hi, 0,1,2,3,4,5,6,7,
                                                 8,9,10,11,12,13,14,15);
            bf16x16 bfS = *(const bf16x16*)(bcolS + kk);
            bf16x16 bfV = *(const bf16x16*)(bcolV + kk);
            accK = wmma_bf16(af, bfS, accK);
            accV = wmma_bf16(af, bfV, accV);
        }
        const float qv    = Q[((size_t)b << 9) + a0 + ln];
        const float biasS = bscore[a0 + ln];
        const float biasV = bvalue[a0 + ln];
        bf16x8 vv;
        #pragma unroll
        for (int v = 0; v < 8; ++v) {
            float kv = accK[v] + biasS;
            kv = kv > 0.f ? kv : 0.f;
            sAcc[v] += kv * qv;                      // scores partial
            float xv = accV[v] + biasV;
            xv = xv > 0.f ? xv : 0.f;
            vv[v] = (__bf16)xv;                      // relu'd value
        }
        // VALS layout [b][a][t]: lane's 8 t's are contiguous -> one b128 store
        *(bf16x8*)(VALS + (((size_t)((b << 9) + a0 + ln)) << 11) + trow) = vv;
    }
    #pragma unroll
    for (int off = 8; off >= 1; off >>= 1)
        #pragma unroll
        for (int v = 0; v < 8; ++v)
            sAcc[v] += __shfl_xor(sAcc[v], off, 32);

    if (ln == 0) {
        #pragma unroll
        for (int v = 0; v < 8; ++v)
            SC[((size_t)b << 11) + trow + v] = sAcc[v];
    }
}

// -----------------------------------------------------------------------------
// Kernel 4: softmax over T with additive -100 mask past listener_len.
// -----------------------------------------------------------------------------
__global__ void softmax_kernel(const float* __restrict__ SC, const int* __restrict__ llen,
                               float* __restrict__ attn) {
    __shared__ float red[256];
    const int b   = blockIdx.x;
    const int tid = threadIdx.x;
    const int L   = llen[b];

    float sv[8];
    float m = -1e30f;
    #pragma unroll
    for (int i = 0; i < 8; ++i) {
        int t = i * 256 + tid;
        float s = SC[((size_t)b << 11) + t] + (t >= L ? -100.0f : 0.0f);
        sv[i] = s;
        m = fmaxf(m, s);
    }
    red[tid] = m; __syncthreads();
    for (int s = 128; s > 0; s >>= 1) {
        if (tid < s) red[tid] = fmaxf(red[tid], red[tid + s]);
        __syncthreads();
    }
    m = red[0]; __syncthreads();

    float sum = 0.f;
    #pragma unroll
    for (int i = 0; i < 8; ++i) { sv[i] = __expf(sv[i] - m); sum += sv[i]; }
    red[tid] = sum; __syncthreads();
    for (int s = 128; s > 0; s >>= 1) {
        if (tid < s) red[tid] += red[tid + s];
        __syncthreads();
    }
    const float inv = 1.0f / red[0];
    #pragma unroll
    for (int i = 0; i < 8; ++i)
        attn[((size_t)b << 11) + i * 256 + tid] = sv[i] * inv;
}

// -----------------------------------------------------------------------------
// FUSED path context: ctx[b][a] = sum_t attn[b][t] * VALS[b][a][t]
// Pure streaming reduction over 64 MB of bf16; one thread per (b,a).
// -----------------------------------------------------------------------------
__global__ void ctx_from_vals_kernel(const __bf16* __restrict__ VALS,
                                     const float* __restrict__ attn,
                                     float* __restrict__ ctxOut) {
    int gid = blockIdx.x * THREADS + threadIdx.x;  // 0 .. 16383
    int b = gid >> 9;
    const __bf16* row = VALS + ((size_t)gid << 11);
    const float*  at  = attn + ((size_t)b << 11);
    float acc = 0.f;
    for (int t = 0; t < T_LEN; t += 8) {
        bf16x8 vv = *(const bf16x8*)(row + t);
        #pragma unroll
        for (int j = 0; j < 8; ++j)
            acc = fmaf(at[t + j], (float)vv[j], acc);
    }
    ctxOut[gid] = acc;
}

// -----------------------------------------------------------------------------
// FALLBACK path (small ws): keys-only scores kernel + recompute-vals context.
// -----------------------------------------------------------------------------
__global__ __launch_bounds__(THREADS)
void scores_kernel(const float* __restrict__ listener, const __bf16* __restrict__ WtS,
                   const float* __restrict__ bscore, const float* __restrict__ Q,
                   float* __restrict__ SC) {
    extern __shared__ char smem[];
    __bf16* As = (__bf16*)smem;

    const int b  = blockIdx.x >> 4;
    const int t0 = (blockIdx.x & 15) * MB;
    stage_A(listener + ((size_t)b << 10), t0, As);
    __syncthreads();

    const int wave = threadIdx.x >> 5;
    const int lane = threadIdx.x & 31;
    const int g    = lane >> 4;
    const int ln   = lane & 15;
    const __bf16* arow = As + (wave * 16 + ln) * LIS_D;

    float sAcc[8] = {0.f, 0.f, 0.f, 0.f, 0.f, 0.f, 0.f, 0.f};
    for (int a0 = 0; a0 < ATT_D; a0 += 16) {
        f32x8 acc = {0.f, 0.f, 0.f, 0.f, 0.f, 0.f, 0.f, 0.f};
        const __bf16* bcol = WtS + ((size_t)(a0 + ln) << 10) + g * 16;
        #pragma unroll 4
        for (int kk = 0; kk < LIS_D; kk += 32) {
            bf16x8 lo = *(const bf16x8*)(arow + kk + g * 16);
            bf16x8 hi = *(const bf16x8*)(arow + kk + g * 16 + 8);
            bf16x16 af = __builtin_shufflevector(lo, hi, 0,1,2,3,4,5,6,7,
                                                 8,9,10,11,12,13,14,15);
            bf16x16 bfr = *(const bf16x16*)(bcol + kk);
            acc = wmma_bf16(af, bfr, acc);
        }
        const float qv   = Q[((size_t)b << 9) + a0 + ln];
        const float bias = bscore[a0 + ln];
        #pragma unroll
        for (int v = 0; v < 8; ++v) {
            float kv = acc[v] + bias;
            kv = kv > 0.f ? kv : 0.f;
            sAcc[v] += kv * qv;
        }
    }
    #pragma unroll
    for (int off = 8; off >= 1; off >>= 1)
        #pragma unroll
        for (int v = 0; v < 8; ++v)
            sAcc[v] += __shfl_xor(sAcc[v], off, 32);
    if (ln == 0) {
        #pragma unroll
        for (int v = 0; v < 8; ++v)
            SC[((size_t)b << 11) + t0 + wave * 16 + g * 8 + v] = sAcc[v];
    }
}

__global__ __launch_bounds__(THREADS)
void context_kernel(const float* __restrict__ listener, const __bf16* __restrict__ WtV,
                    const float* __restrict__ bvalue, const float* __restrict__ attn,
                    float* __restrict__ ctxOut) {
    extern __shared__ char smem[];
    __bf16* As      = (__bf16*)smem;
    float*  ctx_lds = (float*)(smem + MB * LIS_D * sizeof(__bf16));

    const int b  = blockIdx.x >> 4;
    const int t0 = (blockIdx.x & 15) * MB;
    for (int i = threadIdx.x; i < ATT_D; i += THREADS) ctx_lds[i] = 0.0f;
    stage_A(listener + ((size_t)b << 10), t0, As);
    __syncthreads();

    const int wave = threadIdx.x >> 5;
    const int lane = threadIdx.x & 31;
    const int g    = lane >> 4;
    const int ln   = lane & 15;
    const __bf16* arow = As + (wave * 16 + ln) * LIS_D;

    float attnv[8];
    #pragma unroll
    for (int v = 0; v < 8; ++v)
        attnv[v] = attn[((size_t)b << 11) + t0 + wave * 16 + g * 8 + v];

    for (int a0 = 0; a0 < ATT_D; a0 += 16) {
        f32x8 acc = {0.f, 0.f, 0.f, 0.f, 0.f, 0.f, 0.f, 0.f};
        const __bf16* bcol = WtV + ((size_t)(a0 + ln) << 10) + g * 16;
        #pragma unroll 4
        for (int kk = 0; kk < LIS_D; kk += 32) {
            bf16x8 lo = *(const bf16x8*)(arow + kk + g * 16);
            bf16x8 hi = *(const bf16x8*)(arow + kk + g * 16 + 8);
            bf16x16 af = __builtin_shufflevector(lo, hi, 0,1,2,3,4,5,6,7,
                                                 8,9,10,11,12,13,14,15);
            bf16x16 bfr = *(const bf16x16*)(bcol + kk);
            acc = wmma_bf16(af, bfr, acc);
        }
        const float bias = bvalue[a0 + ln];
        float p = 0.f;
        #pragma unroll
        for (int v = 0; v < 8; ++v) {
            float x = acc[v] + bias;
            x = x > 0.f ? x : 0.f;
            p += attnv[v] * x;
        }
        p += __shfl_xor(p, 16, 32);
        if (g == 0) atomicAdd(&ctx_lds[a0 + ln], p);
    }
    __syncthreads();
    for (int i = threadIdx.x; i < ATT_D; i += THREADS)
        unsafeAtomicAdd(&ctxOut[((size_t)b << 9) + i], ctx_lds[i]);
}

// -----------------------------------------------------------------------------
extern "C" void kernel_launch(void* const* d_in, const int* in_sizes, int n_in,
                              void* d_out, int out_size, void* d_ws, size_t ws_size,
                              hipStream_t stream) {
    const float* listener = (const float*)d_in[0];
    const float* speller  = (const float*)d_in[1];
    const int*   llen     = (const int*)d_in[2];
    const float* Ws = (const float*)d_in[4];
    const float* bs = (const float*)d_in[5];
    const float* Wv = (const float*)d_in[6];
    const float* bv = (const float*)d_in[7];
    const float* Wp = (const float*)d_in[8];
    const float* bp = (const float*)d_in[9];

    float* ctxOut  = (float*)d_out;                 // [32, 512]
    float* attnOut = (float*)d_out + B_SZ * ATT_D;  // [32, 2048]

    // workspace layout
    char* ws = (char*)d_ws;
    __bf16* WtS  = (__bf16*)ws;                                  // 1 MB
    __bf16* WtV  = (__bf16*)(ws + (1u << 20));                   // 1 MB
    float*  Q    = (float*)(ws + (2u << 20));                    // 64 KB
    float*  SC   = (float*)(ws + (2u << 20) + (64u << 10));      // 256 KB
    __bf16* VALS = (__bf16*)(ws + (3u << 20));                   // 64 MB (fused)

    const size_t need_fused = (3ull << 20) +
        (size_t)B_SZ * T_LEN * ATT_D * sizeof(__bf16);
    const size_t ldsA = (size_t)MB * LIS_D * sizeof(__bf16);     // 256 KB

    prep_kernel<<<(LIS_D * ATT_D) / THREADS, THREADS, 0, stream>>>(Ws, Wv, WtS, WtV, ctxOut);
    query_kernel<<<B_SZ, ATT_D, 0, stream>>>(speller, Wp, bp, Q);

    if (ws_size >= need_fused) {
        // single pass over x computes scores AND bf16 vals
        scores_vals_kernel<<<B_SZ * (T_LEN / MB), THREADS, ldsA, stream>>>(
            listener, WtS, WtV, bs, bv, Q, SC, VALS);
        softmax_kernel<<<B_SZ, 256, 0, stream>>>(SC, llen, attnOut);
        ctx_from_vals_kernel<<<(B_SZ * ATT_D) / THREADS, THREADS, 0, stream>>>(
            VALS, attnOut, ctxOut);
    } else {
        scores_kernel<<<B_SZ * (T_LEN / MB), THREADS, ldsA, stream>>>(listener, WtS, bs, Q, SC);
        softmax_kernel<<<B_SZ, 256, 0, stream>>>(SC, llen, attnOut);
        context_kernel<<<B_SZ * (T_LEN / MB), THREADS, ldsA + ATT_D * sizeof(float), stream>>>(
            listener, WtV, bv, attnOut, ctxOut);
    }
}
